// QuantumTransformerLayer2025_21638045237787
// MI455X (gfx1250) — matrix operand order repaired and checked by reading, not compiled, not run on hardware
//
#include <hip/hip_runtime.h>

// ---------------- constants (match reference) ----------------
#define BB    4
#define SEQ   1024
#define TTOT  4096          // B*S tokens
#define HIDC  1024
#define NH    8
#define HD    128
#define NQ    8
#define FFC   4096

typedef __bf16 bf16;
typedef __attribute__((ext_vector_type(16))) __bf16 v16bf;
typedef __attribute__((ext_vector_type(8)))  float  v8f;
typedef __attribute__((ext_vector_type(4)))  unsigned int v4u;
typedef __attribute__((ext_vector_type(8)))  int v8i;
typedef __attribute__((ext_vector_type(4)))  int v4i;

#if defined(__has_builtin)
#  if __has_builtin(__builtin_amdgcn_tensor_load_to_lds)
#    define USE_TDM 1
#  endif
#endif
#ifndef USE_TDM
#  define USE_TDM 0
#endif

// ---------------- WMMA helper ----------------
static __device__ __forceinline__ v8f wmma_bf16(v16bf a, v16bf b, v8f c) {
  // (neg_a, A, neg_b, B, c_mod, C, reuse_a, reuse_b)
  return __builtin_amdgcn_wmma_f32_16x16x32_bf16(false, a, false, b, (short)0, c,
                                                 false, false);
}

// A-operand fragment (16x32 bf16): per-lane 16 contiguous K values (32 bytes).
// caller passes p = rowBase + (lane>=16 ? 16 : 0)
static __device__ __forceinline__ v16bf frag_a(const bf16* p) {
  union { v16bf v; uint4 u[2]; } f;
  f.u[0] = *(const uint4*)p;
  f.u[1] = *(const uint4*)(p + 8);
  return f.v;
}
// B-operand fragment (32x16 bf16): per-lane 16 contiguous K values (32 bytes).
static __device__ __forceinline__ v16bf frag_b(const bf16* p) {
  union { v16bf v; uint4 u[2]; } f;
  f.u[0] = *(const uint4*)p;
  f.u[1] = *(const uint4*)(p + 8);
  return f.v;
}

#if USE_TDM
// ---------------- Tensor Data Mover: 2D bf16 tile (global -> LDS) ----------------
// Builds a D# per CDNA5 ISA ch.8: group0 = {count=1, lds_addr, global tile addr, type=2},
// group1 = {data_size=2B, tensor dims, tile dims, dim0 stride}. Groups 2/3 zero (<=2D).
static __device__ __forceinline__ void tdm_load_2d_bf16(
    unsigned lds_addr, const void* gaddr,
    unsigned tensor_d0, unsigned tensor_d1,
    unsigned tile_d0, unsigned tile_d1,
    unsigned long long stride0_elems)
{
  unsigned long long ga = (unsigned long long)gaddr;
  v4u g0;
  g0[0] = 1u;                                            // count=1, user desc
  g0[1] = lds_addr;                                      // LDS byte address
  g0[2] = (unsigned)(ga & 0xffffffffu);                  // global_addr[31:0]
  g0[3] = (unsigned)((ga >> 32) & 0x01ffffffu) | (2u << 30); // addr[56:32] | type=2
  v8i g1;
  g1[0] = (int)(1u << 16);                               // data_size=1 -> 2 bytes
  g1[1] = (int)((tensor_d0 & 0xffffu) << 16);            // tensor_dim0[15:0]
  g1[2] = (int)((tensor_d0 >> 16) | ((tensor_d1 & 0xffffu) << 16));
  g1[3] = (int)((tensor_d1 >> 16) | ((tile_d0 & 0xffffu) << 16));
  g1[4] = (int)(tile_d1 & 0xffffu);                      // tile_dim1 (tile_dim2=0)
  g1[5] = (int)(stride0_elems & 0xffffffffu);            // tensor_dim0_stride[31:0]
  g1[6] = (int)((stride0_elems >> 32) & 0xffffu);        // stride[47:32], dim1_stride=0
  g1[7] = 0;
  v4i z4 = {0, 0, 0, 0};
#if __clang_major__ >= 23
  v8i z8 = {0, 0, 0, 0, 0, 0, 0, 0};
  __builtin_amdgcn_tensor_load_to_lds(g0, g1, z4, z4, z8, 0);
#else
  __builtin_amdgcn_tensor_load_to_lds(g0, g1, z4, z4, 0);
#endif
}
#endif

// ---------------- fp32 -> bf16 conversion ----------------
__global__ void cvt_f32_to_bf16(const float* __restrict__ s, bf16* __restrict__ d, long n) {
  long i = (long)blockIdx.x * blockDim.x + threadIdx.x;
  long stride = (long)gridDim.x * blockDim.x;
  for (; i < n; i += stride) d[i] = (bf16)s[i];
}

// ---------------- generic bf16 GEMM: C = A[M,K] x B[K,N] (+bias) ----------------
// grid = (N/64, M/128, batch). 256 threads = 8 waves; wave -> 32x32 output tile
// (4 accumulators, 4 WMMAs per 32-wide K step, A/B fragments reused twice each).
__global__ __launch_bounds__(256) void gemm_bf16_wmma(
    const bf16* __restrict__ A, int lda,
    const bf16* __restrict__ Bm, int ldb, long bStride,
    const float* __restrict__ bias, int biasStride,
    float* __restrict__ Cf, bf16* __restrict__ Cb, int ldc, long cStride,
    int K)
{
  __shared__ __align__(16) bf16 As[128 * 32];
  __shared__ __align__(16) bf16 Bt[64 * 32];   // [n][k] (transposed tile)

  const int tid = threadIdx.x, lane = tid & 31, w = tid >> 5;
  const int r = lane & 15, hi = lane >> 4;
  const int mt = w & 3, nh = w >> 2;
  const int m0 = blockIdx.y * 128, n0 = blockIdx.x * 64;
  const long z = blockIdx.z;
  const bf16* Bz = Bm + z * bStride;

  v8f acc[2][2];
  acc[0][0] = (v8f){}; acc[0][1] = (v8f){};
  acc[1][0] = (v8f){}; acc[1][1] = (v8f){};

#if USE_TDM
  const unsigned asLds = (unsigned)(size_t)&As[0];
#else
  const int am = tid >> 1, ac = (tid & 1) * 16;    // A stage: 128 rows x 32 cols
#endif
  const int bkr = tid >> 3, bc0 = (tid & 7) * 8;   // B stage: 32 k-rows x 64 cols

  for (int kb = 0; kb < K; kb += 32) {
    __syncthreads();
#if USE_TDM
    if (w == 0)       // TDM ignores EXEC; one issue per workgroup, tracked on TENSORcnt
      tdm_load_2d_bf16(asLds, &A[(size_t)m0 * lda + kb],
                       /*tensor*/ 32, 128, /*tile*/ 32, 128, (unsigned long long)lda);
#else
    *(uint4*)&As[am * 32 + ac]     = *(const uint4*)&A[(size_t)(m0 + am) * lda + kb + ac];
    *(uint4*)&As[am * 32 + ac + 8] = *(const uint4*)&A[(size_t)(m0 + am) * lda + kb + ac + 8];
#endif
    { // B tile: vectorized row load + transposed LDS scatter
      union { bf16 hv[8]; uint4 u; } t2;
      t2.u = *(const uint4*)&Bz[(size_t)(kb + bkr) * ldb + n0 + bc0];
      #pragma unroll
      for (int i = 0; i < 8; i++) Bt[(bc0 + i) * 32 + bkr] = t2.hv[i];
    }
    if (kb + 32 < K)
      __builtin_prefetch(&Bz[(size_t)(kb + 32 + bkr) * ldb + n0 + bc0], 0, 3);
#if USE_TDM
    if (w == 0) __builtin_amdgcn_s_wait_tensorcnt(0);
#endif
    __syncthreads();

    v16bf a0 = frag_a(&As[(mt * 32 + r) * 32 + hi * 16]);
    v16bf a1 = frag_a(&As[(mt * 32 + 16 + r) * 32 + hi * 16]);
    v16bf b0 = frag_b(&Bt[(nh * 32 + r) * 32 + hi * 16]);
    v16bf b1 = frag_b(&Bt[(nh * 32 + 16 + r) * 32 + hi * 16]);
    acc[0][0] = wmma_bf16(a0, b0, acc[0][0]);
    acc[0][1] = wmma_bf16(a0, b1, acc[0][1]);
    acc[1][0] = wmma_bf16(a1, b0, acc[1][0]);
    acc[1][1] = wmma_bf16(a1, b1, acc[1][1]);
  }

  const float* bz = bias ? bias + (size_t)z * biasStride : nullptr;
  float* cfz = Cf ? Cf + z * cStride : nullptr;
  bf16*  cbz = Cb ? Cb + z * cStride : nullptr;
  #pragma unroll
  for (int sm = 0; sm < 2; sm++) {
    #pragma unroll
    for (int i = 0; i < 8; i++) {
      int row = m0 + mt * 32 + sm * 16 + i + hi * 8;
      #pragma unroll
      for (int sn = 0; sn < 2; sn++) {
        int c = n0 + nh * 32 + sn * 16 + r;
        float v = acc[sm][sn][i];
        if (bz) v += bz[c];
        if (cfz) cfz[(size_t)row * ldc + c] = v;
        if (cbz) cbz[(size_t)row * ldc + c] = (bf16)v;
      }
    }
  }
}

// ---------------- classical->quantum projection + norm ----------------
__global__ void c2q_kernel(const bf16* __restrict__ Xh, const float* __restrict__ Wc2q,
                           const float* __restrict__ bc2q,
                           float* __restrict__ outq, float* __restrict__ outn)
{
  int idx = blockIdx.x * 256 + threadIdx.x;
  if (idx >= NH * TTOT) return;
  int h = idx / TTOT;
  const bf16* xr = Xh + (size_t)idx * HD;   // layout [H][T][HD]
  const float* W = Wc2q + h * HD * NQ;
  float acc[NQ];
  #pragma unroll
  for (int j = 0; j < NQ; j++) acc[j] = bc2q[h * NQ + j];
  for (int d = 0; d < HD; d++) {
    float xv = (float)xr[d];
    #pragma unroll
    for (int j = 0; j < NQ; j++) acc[j] += xv * W[d * NQ + j];
  }
  float s = 0.f;
  #pragma unroll
  for (int j = 0; j < NQ; j++) { outq[(size_t)idx * NQ + j] = acc[j]; s += acc[j] * acc[j]; }
  outn[idx] = sqrtf(s);
}

// ---------------- flash attention with quantum/classical mixed scores ----------------
// grid = (SEQ/128, NH, BB); 256 threads, wave w owns 16 query rows.
__global__ __launch_bounds__(256) void attention_kernel(
    const bf16* __restrict__ Qb, const bf16* __restrict__ Kb, const bf16* __restrict__ Vb,
    const float* __restrict__ qq, const float* __restrict__ qn,
    const float* __restrict__ kq, const float* __restrict__ kn,
    bf16* __restrict__ heads)
{
  __shared__ __align__(16) bf16 Ks[32 * HD];
  __shared__ __align__(16) bf16 Vt[HD * 32];      // transposed V block
  __shared__ __align__(16) bf16 Pb[8][16 * 32];   // per-wave P relayout buffer
  __shared__ float kqs[32 * NQ];
  __shared__ float kns[32];
  __shared__ float qqs[128 * NQ];
  __shared__ float qns[128];

  const int tid = threadIdx.x, lane = tid & 31, w = tid >> 5;
  const int r = lane & 15, hi = lane >> 4;
  const int h = blockIdx.y, bb = blockIdx.z;
  const int tb = bb * SEQ;
  const int q0 = blockIdx.x * 128;

  const bf16* Qh = Qb + (size_t)h * TTOT * HD;
  const bf16* Kh = Kb + (size_t)h * TTOT * HD;
  const bf16* Vh = Vb + (size_t)h * TTOT * HD;
  const float* qqh = qq + (size_t)h * TTOT * NQ;
  const float* kqh = kq + (size_t)h * TTOT * NQ;
  const float* qnh = qn + (size_t)h * TTOT;
  const float* knh = kn + (size_t)h * TTOT;

#if USE_TDM
  const unsigned ksLds = (unsigned)(size_t)&Ks[0];
#endif

  // resident Q A-fragments (16 rows x HD=128, 4 chunks of K=32)
  v16bf qf[4];
  {
    const bf16* qr = Qh + (size_t)(tb + q0 + w * 16 + r) * HD;
    #pragma unroll
    for (int kc = 0; kc < 4; kc++) qf[kc] = frag_a(qr + kc * 32 + hi * 16);
  }
  for (int i = tid; i < 128; i += 256) {
    int g = tb + q0 + i;
    #pragma unroll
    for (int j = 0; j < NQ; j++) qqs[i * NQ + j] = qqh[(size_t)g * NQ + j];
    qns[i] = qnh[g];
  }

  float m[8], l[8];
  v8f o[8];
  #pragma unroll
  for (int i = 0; i < 8; i++) { m[i] = -1e30f; l[i] = 0.f; o[i] = (v8f){}; }

  const float invs = 0.0883883476483184f;         // 1/sqrt(128)
  const float PIf  = 3.14159265358979323846f;

  for (int kb = 0; kb < SEQ; kb += 32) {
    __syncthreads();
#if USE_TDM
    // K block rows are consecutive tokens -> one contiguous 8KB TDM transfer.
    if (w == 0)
      tdm_load_2d_bf16(ksLds, Kh + (size_t)(tb + kb) * HD,
                       /*tensor*/ 32 * HD, 1, /*tile*/ 32 * HD, 1, (unsigned long long)(32 * HD));
#else
    { // stage K block (row-major): B-operand of QK^T reads key rows directly
      int row = tid >> 3, c0 = (tid & 7) * 16;
      const bf16* src = Kh + (size_t)(tb + kb + row) * HD + c0;
      *(uint4*)&Ks[row * HD + c0]     = *(const uint4*)src;
      *(uint4*)&Ks[row * HD + c0 + 8] = *(const uint4*)(src + 8);
    }
#endif
    { // stage V transposed: Vt[d][key]
      int key = tid & 31, d0 = (tid >> 5) * 16;
      union { bf16 hv[16]; uint4 u[2]; } t2;
      const bf16* src = Vh + (size_t)(tb + kb + key) * HD + d0;
      t2.u[0] = *(const uint4*)src; t2.u[1] = *(const uint4*)(src + 8);
      #pragma unroll
      for (int i = 0; i < 16; i++) Vt[(d0 + i) * 32 + key] = t2.hv[i];
    }
    if (tid < 32) {
      int g = tb + kb + tid;
      #pragma unroll
      for (int j = 0; j < NQ; j++) kqs[tid * NQ + j] = kqh[(size_t)g * NQ + j];
      kns[tid] = knh[g];
    }
#if USE_TDM
    if (w == 0) __builtin_amdgcn_s_wait_tensorcnt(0);
#endif
    __syncthreads();

    // scores: two 16x16 fragments (keys kb..+15, kb+16..+31)
    v8f s0 = {}, s1 = {};
    #pragma unroll
    for (int kc = 0; kc < 4; kc++) {
      v16bf bk0 = frag_b(&Ks[r * HD        + kc * 32 + hi * 16]);
      v16bf bk1 = frag_b(&Ks[(16 + r) * HD + kc * 32 + hi * 16]);
      s0 = wmma_bf16(qf[kc], bk0, s0);
      s1 = wmma_bf16(qf[kc], bk1, s1);
    }

    float kq0[NQ], kq1[NQ];
    #pragma unroll
    for (int j = 0; j < NQ; j++) { kq0[j] = kqs[r * NQ + j]; kq1[j] = kqs[(r + 16) * NQ + j]; }
    float kn0 = kns[r], kn1 = kns[r + 16];

    #pragma unroll
    for (int i = 0; i < 8; i++) {
      int lr = w * 16 + i + hi * 8;                 // lane's query row (block-local)
      float d0 = 0.f, d1 = 0.f;
      #pragma unroll
      for (int j = 0; j < NQ; j++) { float qv = qqs[lr * NQ + j]; d0 += qv * kq0[j]; d1 += qv * kq1[j]; }
      float qni = qns[lr];
      float fid0 = d0 / fmaxf(qni * kn0, 1e-8f);
      float fid1 = d1 / fmaxf(qni * kn1, 1e-8f);
      float qw0 = fid0 * (1.f + 0.1f * __sinf(fid0 * PIf));
      float qw1 = fid1 * (1.f + 0.1f * __sinf(fid1 * PIf));
      float aw0 = 0.5f * qw0 + 0.5f * (s0[i] * invs);
      float aw1 = 0.5f * qw1 + 0.5f * (s1[i] * invs);

      // online softmax over this row (16 cols live in one 16-lane half)
      float mx = fmaxf(aw0, aw1);
      #pragma unroll
      for (int msk = 1; msk <= 8; msk <<= 1) mx = fmaxf(mx, __shfl_xor(mx, msk, 32));
      float mn = fmaxf(m[i], mx);
      float sc = __expf(m[i] - mn);
      m[i] = mn;
      float p0 = __expf(aw0 - mn), p1 = __expf(aw1 - mn);
      float rs = p0 + p1;
      #pragma unroll
      for (int msk = 1; msk <= 8; msk <<= 1) rs += __shfl_xor(rs, msk, 32);
      l[i] = l[i] * sc + rs;
      #pragma unroll
      for (int dt = 0; dt < 8; dt++) o[dt][i] *= sc;
      Pb[w][(i + hi * 8) * 32 + r]      = (bf16)p0;
      Pb[w][(i + hi * 8) * 32 + 16 + r] = (bf16)p1;
    }

    // P (16x32) @ V (32x128): A-fragment from LDS relayout, B from Vt
    v16bf pf = frag_a(&Pb[w][r * 32 + hi * 16]);
    #pragma unroll
    for (int dt = 0; dt < 8; dt++) {
      v16bf vf = frag_b(&Vt[(dt * 16 + r) * 32 + hi * 16]);
      o[dt] = wmma_bf16(pf, vf, o[dt]);
    }
  }

  #pragma unroll
  for (int i = 0; i < 8; i++) l[i] = 1.f / l[i];
  #pragma unroll
  for (int dt = 0; dt < 8; dt++) {
    #pragma unroll
    for (int i = 0; i < 8; i++) {
      int row = tb + q0 + w * 16 + i + hi * 8;
      int col = h * HD + dt * 16 + r;
      heads[(size_t)row * HIDC + col] = (bf16)(o[dt][i] * l[i]);
    }
  }
}

// ---------------- fused residual-add + LayerNorm ----------------
__global__ __launch_bounds__(256) void add_ln(
    const float* __restrict__ xa, const float* __restrict__ yb,
    const float* __restrict__ g, const float* __restrict__ be,
    float* __restrict__ outF, bf16* __restrict__ outB)
{
  const int t = blockIdx.x, tid = threadIdx.x;
  float v4[4]; float s = 0.f, ss = 0.f;
  #pragma unroll
  for (int i = 0; i < 4; i++) {
    int c = tid + i * 256;
    float v = xa[(size_t)t * HIDC + c] + yb[(size_t)t * HIDC + c];
    v4[i] = v; s += v; ss += v * v;
  }
  #pragma unroll
  for (int msk = 16; msk >= 1; msk >>= 1) { s += __shfl_xor(s, msk, 32); ss += __shfl_xor(ss, msk, 32); }
  __shared__ float rs[8], rss[8];
  if ((tid & 31) == 0) { rs[tid >> 5] = s; rss[tid >> 5] = ss; }
  __syncthreads();
  float S = 0.f, SS = 0.f;
  #pragma unroll
  for (int i = 0; i < 8; i++) { S += rs[i]; SS += rss[i]; }
  float mean = S * (1.f / HIDC);
  float var  = SS * (1.f / HIDC) - mean * mean;
  float inv  = rsqrtf(var + 1e-5f);
  #pragma unroll
  for (int i = 0; i < 4; i++) {
    int c = tid + i * 256;
    float y = (v4[i] - mean) * inv * g[c] + be[c];
    outF[(size_t)t * HIDC + c] = y;
    if (outB) outB[(size_t)t * HIDC + c] = (bf16)y;
  }
}

// ---------------- qm = h@Wm+bm ; qe = tanh(qm)*cos(qm*pi/4) ----------------
__global__ __launch_bounds__(256) void qff_mid(
    const bf16* __restrict__ hbuf, const float* __restrict__ Wm,
    const float* __restrict__ bm, float* __restrict__ qe)
{
  const int t = blockIdx.x, tid = threadIdx.x;
  float acc[NQ] = {};
  for (int f = tid; f < FFC; f += 256) {
    float hv = (float)hbuf[(size_t)t * FFC + f];
    #pragma unroll
    for (int j = 0; j < NQ; j++) acc[j] += hv * Wm[f * NQ + j];
  }
  __shared__ float red[256];
  __shared__ float qmsh[NQ];
  for (int j = 0; j < NQ; j++) {
    red[tid] = acc[j];
    __syncthreads();
    for (int s2 = 128; s2 > 0; s2 >>= 1) {
      if (tid < s2) red[tid] += red[tid + s2];
      __syncthreads();
    }
    if (tid == 0) qmsh[j] = red[0] + bm[j];
    __syncthreads();
  }
  if (tid < NQ) {
    float v = qmsh[tid];
    qe[(size_t)t * NQ + tid] = tanhf(v) * __cosf(v * 0.78539816339744831f);
  }
}

// ---------------- ha = qe @ Wqo + bqo (rank-8 expansion) ----------------
__global__ void qff_out(const float* __restrict__ qe, const float* __restrict__ Wqo,
                        const float* __restrict__ bqo, bf16* __restrict__ hab)
{
  long n = (long)TTOT * FFC;
  long i = (long)blockIdx.x * 256 + threadIdx.x;
  long stride = (long)gridDim.x * 256;
  for (; i < n; i += stride) {
    int t = (int)(i >> 12);       // FFC == 4096
    int f = (int)(i & 4095);
    float a = bqo[f];
    #pragma unroll
    for (int j = 0; j < NQ; j++) a += qe[(size_t)t * NQ + j] * Wqo[(size_t)j * FFC + f];
    hab[i] = (bf16)a;
  }
}

// =======================================================================
extern "C" void kernel_launch(void* const* d_in, const int* in_sizes, int n_in,
                              void* d_out, int out_size, void* d_ws, size_t ws_size,
                              hipStream_t stream) {
  (void)in_sizes; (void)n_in; (void)out_size; (void)ws_size;
  const float* x     = (const float*)d_in[0];
  const float* Wq    = (const float*)d_in[1];
  const float* bq    = (const float*)d_in[2];
  const float* Wk    = (const float*)d_in[3];
  const float* bk    = (const float*)d_in[4];
  const float* Wv    = (const float*)d_in[5];
  const float* bv    = (const float*)d_in[6];
  const float* Wc2q  = (const float*)d_in[7];
  const float* bc2q  = (const float*)d_in[8];
  const float* Wo    = (const float*)d_in[9];
  const float* bo    = (const float*)d_in[10];
  const float* g1    = (const float*)d_in[11];
  const float* beta1 = (const float*)d_in[12];
  const float* g2    = (const float*)d_in[13];
  const float* beta2 = (const float*)d_in[14];
  const float* W1    = (const float*)d_in[15];
  const float* b1    = (const float*)d_in[16];
  const float* Wm    = (const float*)d_in[17];
  const float* bm    = (const float*)d_in[18];
  const float* Wqo   = (const float*)d_in[19];
  const float* bqo   = (const float*)d_in[20];
  const float* W2    = (const float*)d_in[21];
  const float* b2    = (const float*)d_in[22];
  float* out = (float*)d_out;

  // ---- workspace carve-up (256B aligned) ----
  char* ws = (char*)d_ws;
  size_t off = 0;
  auto take = [&](size_t bytes) -> char* {
    char* p = ws + off;
    off += (bytes + 255) & ~(size_t)255;
    return p;
  };
  bf16* xb   = (bf16*)take((size_t)TTOT * HIDC * 2);
  bf16* Wqb  = (bf16*)take((size_t)NH * HIDC * HD * 2);
  bf16* Wkb  = (bf16*)take((size_t)NH * HIDC * HD * 2);
  bf16* Wvb  = (bf16*)take((size_t)NH * HIDC * HD * 2);
  bf16* Wob  = (bf16*)take((size_t)HIDC * HIDC * 2);
  bf16* W1b  = (bf16*)take((size_t)HIDC * FFC * 2);
  bf16* W2b  = (bf16*)take((size_t)FFC * HIDC * 2);
  bf16* Qb   = (bf16*)take((size_t)NH * TTOT * HD * 2);
  bf16* Kbq  = (bf16*)take((size_t)NH * TTOT * HD * 2);
  bf16* Vb   = (bf16*)take((size_t)NH * TTOT * HD * 2);
  float* qqB = (float*)take((size_t)NH * TTOT * NQ * 4);
  float* kqB = (float*)take((size_t)NH * TTOT * NQ * 4);
  float* qnB = (float*)take((size_t)NH * TTOT * 4);
  float* knB = (float*)take((size_t)NH * TTOT * 4);
  bf16* headsb = (bf16*)take((size_t)TTOT * HIDC * 2);
  float* attnf = (float*)take((size_t)TTOT * HIDC * 4);
  float* x1f   = (float*)take((size_t)TTOT * HIDC * 4);
  bf16*  x1b   = (bf16*)take((size_t)TTOT * HIDC * 2);
  bf16*  hbuf  = (bf16*)take((size_t)TTOT * FFC * 2);
  float* qeB   = (float*)take((size_t)TTOT * NQ * 4);
  bf16*  hab   = (bf16*)take((size_t)TTOT * FFC * 2);
  float* ffb   = (float*)take((size_t)TTOT * HIDC * 4);

  dim3 blk(256);

  // ---- fp32 -> bf16 conversions ----
  cvt_f32_to_bf16<<<4096, blk, 0, stream>>>(x,  xb,  (long)TTOT * HIDC);
  cvt_f32_to_bf16<<<2048, blk, 0, stream>>>(Wq, Wqb, (long)NH * HIDC * HD);
  cvt_f32_to_bf16<<<2048, blk, 0, stream>>>(Wk, Wkb, (long)NH * HIDC * HD);
  cvt_f32_to_bf16<<<2048, blk, 0, stream>>>(Wv, Wvb, (long)NH * HIDC * HD);
  cvt_f32_to_bf16<<<2048, blk, 0, stream>>>(Wo, Wob, (long)HIDC * HIDC);
  cvt_f32_to_bf16<<<4096, blk, 0, stream>>>(W1, W1b, (long)HIDC * FFC);
  cvt_f32_to_bf16<<<4096, blk, 0, stream>>>(W2, W2b, (long)FFC * HIDC);

  // ---- QKV projections: batched over heads via grid.z ----
  gemm_bf16_wmma<<<dim3(HD / 64, TTOT / 128, NH), blk, 0, stream>>>(
      xb, HIDC, Wqb, HD, (long)HIDC * HD, bq, HD, nullptr, Qb, HD, (long)TTOT * HD, HIDC);
  gemm_bf16_wmma<<<dim3(HD / 64, TTOT / 128, NH), blk, 0, stream>>>(
      xb, HIDC, Wkb, HD, (long)HIDC * HD, bk, HD, nullptr, Kbq, HD, (long)TTOT * HD, HIDC);
  gemm_bf16_wmma<<<dim3(HD / 64, TTOT / 128, NH), blk, 0, stream>>>(
      xb, HIDC, Wvb, HD, (long)HIDC * HD, bv, HD, nullptr, Vb, HD, (long)TTOT * HD, HIDC);

  // ---- quantum projections + norms ----
  c2q_kernel<<<(NH * TTOT + 255) / 256, blk, 0, stream>>>(Qb,  Wc2q, bc2q, qqB, qnB);
  c2q_kernel<<<(NH * TTOT + 255) / 256, blk, 0, stream>>>(Kbq, Wc2q, bc2q, kqB, knB);

  // ---- attention ----
  attention_kernel<<<dim3(SEQ / 128, NH, BB), blk, 0, stream>>>(
      Qb, Kbq, Vb, qqB, qnB, kqB, knB, headsb);

  // ---- output projection ----
  gemm_bf16_wmma<<<dim3(HIDC / 64, TTOT / 128, 1), blk, 0, stream>>>(
      headsb, HIDC, Wob, HIDC, 0, bo, 0, attnf, nullptr, HIDC, 0, HIDC);

  // ---- LN1 ----
  add_ln<<<TTOT, blk, 0, stream>>>(x, attnf, g1, beta1, x1f, x1b);

  // ---- FFN up ----
  gemm_bf16_wmma<<<dim3(FFC / 64, TTOT / 128, 1), blk, 0, stream>>>(
      x1b, HIDC, W1b, FFC, 0, b1, 0, nullptr, hbuf, FFC, 0, HIDC);

  // ---- quantum middle ----
  qff_mid<<<TTOT, blk, 0, stream>>>(hbuf, Wm, bm, qeB);
  qff_out<<<8192, blk, 0, stream>>>(qeB, Wqo, bqo, hab);

  // ---- FFN down ----
  gemm_bf16_wmma<<<dim3(HIDC / 64, TTOT / 128, 1), blk, 0, stream>>>(
      hab, FFC, W2b, HIDC, 0, b2, 0, ffb, nullptr, HIDC, 0, FFC);

  // ---- LN2 -> output ----
  add_ln<<<TTOT, blk, 0, stream>>>(x1f, ffb, g2, beta2, out, nullptr);
}
